// TimeEmbedder_24292335026674
// MI455X (gfx1250) — matrix operand order repaired and checked by reading, not compiled, not run on hardware
//
#include <hip/hip_runtime.h>
#include <math.h>

typedef __attribute__((ext_vector_type(16))) __bf16        v16bf;
typedef __attribute__((ext_vector_type(8)))  float         v8f;
typedef __attribute__((ext_vector_type(4)))  unsigned int  u32x4;

#define TE_DIM   1024
#define TE_HID   4096
#define TE_BATCH 32768

union FragBf { v16bf v; u32x4 u[2]; };

// ---------------------------------------------------------------------------
// Kernel 1: EDM positional embedding + per-row normalization (ddof=1), bf16 out
// One 512-thread block per row: thread j owns (cos_j, sin_j) -> cols j, j+512.
// ---------------------------------------------------------------------------
__global__ __launch_bounds__(512)
void te_embed_norm(const float* __restrict__ t, __bf16* __restrict__ Tn) {
    const int row = blockIdx.x;
    const int j   = threadIdx.x;                 // 0..511
    const float x  = t[row] * 1000.0f;           // TIME_SCALING
    // freq = (1/10000)^(j/511)  (endpoint=True -> divisor half-1)
    const float fr = __expf((float)j * (1.0f / 511.0f) * -9.210340371976184f);
    const float ang = x * fr;
    float s, c;
    sincosf(ang, &s, &c);

    __shared__ float red[512];
    red[j] = c + s;
    __syncthreads();
    for (int off = 256; off > 0; off >>= 1) {
        if (j < off) red[j] += red[j + off];
        __syncthreads();
    }
    const float mean = red[0] * (1.0f / 1024.0f);
    __syncthreads();
    const float dc = c - mean, ds = s - mean;
    red[j] = dc * dc + ds * ds;
    __syncthreads();
    for (int off = 256; off > 0; off >>= 1) {
        if (j < off) red[j] += red[j + off];
        __syncthreads();
    }
    const float rstd = rsqrtf(red[0] * (1.0f / 1023.0f));   // unbiased (ddof=1)

    __bf16* o = Tn + (size_t)row * TE_DIM;
    o[j]       = (__bf16)(dc * rstd);
    o[j + 512] = (__bf16)(ds * rstd);
}

// ---------------------------------------------------------------------------
// Kernel 2: f32 [K][N] -> bf16 [N][K] transpose-convert (32x32 LDS tiles)
// ---------------------------------------------------------------------------
__global__ __launch_bounds__(256)
void te_transpose_bf16(const float* __restrict__ in, __bf16* __restrict__ out,
                       int K, int N) {
    __shared__ float tile[32][33];
    const int tx = threadIdx.x;                  // 0..31
    const int ty = threadIdx.y;                  // 0..7
    const int n0 = blockIdx.x * 32;
    const int k0 = blockIdx.y * 32;
#pragma unroll
    for (int jj = 0; jj < 4; ++jj)
        tile[ty + jj * 8][tx] = in[(size_t)(k0 + ty + jj * 8) * N + n0 + tx];
    __syncthreads();
#pragma unroll
    for (int jj = 0; jj < 4; ++jj)
        out[(size_t)(n0 + ty + jj * 8) * K + k0 + tx] = (__bf16)tile[tx][ty + jj * 8];
}

// ---------------------------------------------------------------------------
// Kernel 3: bf16 WMMA GEMM  C[M][N] = A[M][K] * Bt[N][K]^T + bias (+ SiLU)
// Block: 256 threads (8 waves), tile 128x128x32, double-buffered LDS.
// Wave (wm in 0..1, wn in 0..3) owns a 64x32 sub-tile = 4x2 WMMA 16x16 tiles.
// Pipeline per iteration: global_load(k+1) -> 12x ds_load frags(k) ->
// 8x v_wmma burst -> ds_store(k+1) -> one barrier.
// ---------------------------------------------------------------------------
template <bool SILU, typename OutT>
__global__ __launch_bounds__(256)
void te_gemm_wmma(const __bf16* __restrict__ A,    // [M][K] bf16 row-major
                  const __bf16* __restrict__ Bt,   // [N][K] bf16 row-major
                  const float*  __restrict__ bias, // [N]
                  OutT* __restrict__ C,            // [M][N]
                  int M, int N, int K) {
    constexpr int BM = 128, BN = 128, BK = 32;
    // +8 bf16 pad -> 80-byte row stride: 16B-aligned and bank-conflict-free
    __shared__ __bf16 As[2][BM][BK + 8];
    __shared__ __bf16 Bs[2][BN][BK + 8];

    const int tid   = threadIdx.x;
    const int lane  = tid & 31;
    const int wave  = tid >> 5;          // 0..7
    const int wm    = wave >> 2;         // 0..1 : 64-row slab
    const int wn    = wave & 3;          // 0..3 : 32-col slab
    const int lhalf = lane >> 4;         // 0/1
    const int l15   = lane & 15;
    const int kbA   = lhalf * 8;         // A frag K chunk base (ISA 16-bit A layout)
    const int kbB   = lhalf * 8;         // B frag K chunk base (same K order as A)

    const int rowBase = blockIdx.y * BM;
    const int colBase = blockIdx.x * BN;

    v8f acc[4][2];
#pragma unroll
    for (int mt = 0; mt < 4; ++mt)
#pragma unroll
        for (int nt = 0; nt < 2; ++nt)
#pragma unroll
            for (int e = 0; e < 8; ++e) acc[mt][nt][e] = 0.0f;

    // cooperative staging: 512 16-byte chunks per tile, 2 per thread per matrix
    const int lr = tid >> 2;             // 0..63 : row within half-tile
    const int lc = tid & 3;              // 16B chunk within 64B row

    const __bf16* Abase = A  + (size_t)(rowBase + lr) * K + lc * 8;
    const __bf16* Bbase = Bt + (size_t)(colBase + lr) * K + lc * 8;
    const size_t  strideHalf = (size_t)64 * K;   // 64 rows down

    // ---- prologue: stage tile 0 into buffer 0 ----
    {
        const u32x4 a0 = *reinterpret_cast<const u32x4*>(Abase);
        const u32x4 a1 = *reinterpret_cast<const u32x4*>(Abase + strideHalf);
        const u32x4 b0 = *reinterpret_cast<const u32x4*>(Bbase);
        const u32x4 b1 = *reinterpret_cast<const u32x4*>(Bbase + strideHalf);
        *reinterpret_cast<u32x4*>(&As[0][lr][lc * 8])      = a0;
        *reinterpret_cast<u32x4*>(&As[0][lr + 64][lc * 8]) = a1;
        *reinterpret_cast<u32x4*>(&Bs[0][lr][lc * 8])      = b0;
        *reinterpret_cast<u32x4*>(&Bs[0][lr + 64][lc * 8]) = b1;
    }
    __syncthreads();

    const int ntiles = K / BK;
    for (int it = 0; it < ntiles; ++it) {
        const int  buf  = it & 1;
        const bool more = (it + 1) < ntiles;

        // issue global loads for tile it+1 (latency hidden by WMMA burst below)
        u32x4 a0, a1, b0, b1;
        if (more) {
            const __bf16* Ag = Abase + (size_t)(it + 1) * BK;
            const __bf16* Bg = Bbase + (size_t)(it + 1) * BK;
            a0 = *reinterpret_cast<const u32x4*>(Ag);
            a1 = *reinterpret_cast<const u32x4*>(Ag + strideHalf);
            b0 = *reinterpret_cast<const u32x4*>(Bg);
            b1 = *reinterpret_cast<const u32x4*>(Bg + strideHalf);
        }
        if (it + 2 < ntiles) {           // warm L2 for the slice after next
            __builtin_prefetch(Abase + (size_t)(it + 2) * BK, 0, 1);
            __builtin_prefetch(Bbase + (size_t)(it + 2) * BK, 0, 1);
        }

        // load ALL fragments up front -> partial dscnt waits, dense WMMA burst
        FragBf fa[4], fb[2];
#pragma unroll
        for (int nt = 0; nt < 2; ++nt) {
            const int n = wn * 32 + nt * 16 + l15;   // col = lane&15
            fb[nt].u[0] = *reinterpret_cast<const u32x4*>(&Bs[buf][n][kbB]);
            fb[nt].u[1] = *reinterpret_cast<const u32x4*>(&Bs[buf][n][kbB + 16]);
        }
#pragma unroll
        for (int mt = 0; mt < 4; ++mt) {
            const int r = wm * 64 + mt * 16 + l15;   // row = lane&15
            fa[mt].u[0] = *reinterpret_cast<const u32x4*>(&As[buf][r][kbA]);
            fa[mt].u[1] = *reinterpret_cast<const u32x4*>(&As[buf][r][kbA + 16]);
        }
#pragma unroll
        for (int mt = 0; mt < 4; ++mt)
#pragma unroll
            for (int nt = 0; nt < 2; ++nt)
                acc[mt][nt] = __builtin_amdgcn_wmma_f32_16x16x32_bf16(
                    false, fa[mt].v, false, fb[nt].v,
                    (short)0, acc[mt][nt], false, false);

        // stash tile it+1 into the other buffer (safe: buf^1 last read at it-1,
        // protected by the barrier at the end of iteration it-1)
        if (more) {
            const int nb = buf ^ 1;
            *reinterpret_cast<u32x4*>(&As[nb][lr][lc * 8])      = a0;
            *reinterpret_cast<u32x4*>(&As[nb][lr + 64][lc * 8]) = a1;
            *reinterpret_cast<u32x4*>(&Bs[nb][lr][lc * 8])      = b0;
            *reinterpret_cast<u32x4*>(&Bs[nb][lr + 64][lc * 8]) = b1;
        }
        __syncthreads();
    }

    // Epilogue: C VGPR e holds (M = e + 8*lhalf, N = lane&15) per ISA C/D layout
#pragma unroll
    for (int mt = 0; mt < 4; ++mt) {
#pragma unroll
        for (int nt = 0; nt < 2; ++nt) {
            const int col = colBase + wn * 32 + nt * 16 + l15;
            const float b = bias[col];
#pragma unroll
            for (int e = 0; e < 8; ++e) {
                const int row = rowBase + wm * 64 + mt * 16 + e + lhalf * 8;
                float v = acc[mt][nt][e] + b;
                if (SILU) v = v / (1.0f + __expf(-v));
                C[(size_t)row * N + col] = (OutT)v;
            }
        }
    }
}

// ---------------------------------------------------------------------------
// Launcher
// inputs: 0=t [32768] f32, 1=W1 [1024,4096] f32, 2=b1 [4096] f32,
//         3=W2 [4096,4096] f32, 4=b2 [4096] f32 ; out: [32768,4096] f32
// ws layout: Tn(64MB) | W1t(8MB) | W2t(32MB) | H(256MB)  = 360 MB
// ---------------------------------------------------------------------------
extern "C" void kernel_launch(void* const* d_in, const int* in_sizes, int n_in,
                              void* d_out, int out_size, void* d_ws, size_t ws_size,
                              hipStream_t stream) {
    const float* t  = (const float*)d_in[0];
    const float* W1 = (const float*)d_in[1];
    const float* b1 = (const float*)d_in[2];
    const float* W2 = (const float*)d_in[3];
    const float* b2 = (const float*)d_in[4];
    float* out = (float*)d_out;

    char* ws = (char*)d_ws;
    __bf16* Tn  = (__bf16*)ws;  ws += (size_t)TE_BATCH * TE_DIM * sizeof(__bf16);
    __bf16* W1t = (__bf16*)ws;  ws += (size_t)TE_DIM   * TE_HID * sizeof(__bf16);
    __bf16* W2t = (__bf16*)ws;  ws += (size_t)TE_HID   * TE_HID * sizeof(__bf16);
    __bf16* H   = (__bf16*)ws;

    // 1) embedding + normalize -> bf16
    te_embed_norm<<<TE_BATCH, 512, 0, stream>>>(t, Tn);

    // 2) weights f32 -> bf16, transposed to [N][K]
    te_transpose_bf16<<<dim3(TE_HID / 32, TE_DIM / 32), dim3(32, 8), 0, stream>>>(
        W1, W1t, TE_DIM, TE_HID);
    te_transpose_bf16<<<dim3(TE_HID / 32, TE_HID / 32), dim3(32, 8), 0, stream>>>(
        W2, W2t, TE_HID, TE_HID);

    // 3) H = silu(Tn @ W1 + b1)   [32768 x 4096] bf16
    te_gemm_wmma<true, __bf16><<<dim3(TE_HID / 128, TE_BATCH / 128), 256, 0, stream>>>(
        Tn, W1t, b1, H, TE_BATCH, TE_HID, TE_DIM);

    // 4) out = H @ W2 + b2        [32768 x 4096] f32
    te_gemm_wmma<false, float><<<dim3(TE_HID / 128, TE_BATCH / 128), 256, 0, stream>>>(
        H, W2t, b2, out, TE_BATCH, TE_HID, TE_HID);
}
